// MoE_88510686035995
// MI455X (gfx1250) — compile-verified, matrix operand order checked by reading, not run on hardware
//
#include <hip/hip_runtime.h>
#include <hip/hip_bf16.h>

// ---------------------------------------------------------------------------
// Types for CDNA5 WMMA (wave32, 16x16x32 bf16 -> f32) and TDM descriptors
// ---------------------------------------------------------------------------
typedef __attribute__((ext_vector_type(16))) __bf16       v16bf;
typedef __attribute__((ext_vector_type(8)))  float        v8f;
typedef __attribute__((ext_vector_type(4)))  float        f32x4;
typedef __attribute__((ext_vector_type(4)))  unsigned int u32x4;
typedef __attribute__((ext_vector_type(4)))  int          i32x4;
typedef __attribute__((ext_vector_type(8)))  int          i32x8;

union AF { v16bf v; u32x4 q[2]; };

__device__ __forceinline__ unsigned short bf16_bits(float f) {
    union { float f; unsigned u; } c; c.f = f;
    unsigned u = c.u + 0x7FFFu + ((c.u >> 16) & 1u);   // round-to-nearest-even
    return (unsigned short)(u >> 16);
}
__device__ __forceinline__ unsigned pack2(float a, float b) {
    return (unsigned)bf16_bits(a) | ((unsigned)bf16_bits(b) << 16);
}
__device__ __forceinline__ v8f v8f_zero() {
    v8f z;
#pragma unroll
    for (int i = 0; i < 8; ++i) z[i] = 0.0f;
    return z;
}

// ---------------------------------------------------------------------------
// TDM: 2D bf16 tile load global->LDS.
//   tdim0/tdim1: tensor extents (OOB rows/cols zero-filled)
//   tile_d0/tile_d1: tile extents in elements
//   stride: tensor row stride in elements
//   cfg: group1 word0 (data_size | pad_enable | pad_interval | pad_amount)
// ---------------------------------------------------------------------------
#if __has_builtin(__builtin_amdgcn_tensor_load_to_lds)
#define USE_TDM 1
__device__ __forceinline__ void tdm_load_tile(unsigned lds_off, const void* gptr,
                                              int tdim0, int tdim1,
                                              int tile_d0, int tile_d1,
                                              long stride, unsigned cfg)
{
    unsigned long long ga = (unsigned long long)(uintptr_t)gptr;
    u32x4 g0;
    g0[0] = 1u;                                   // count=1, user descriptor
    g0[1] = lds_off;                              // lds_addr
    g0[2] = (unsigned)ga;                         // global_addr[31:0]
    g0[3] = (unsigned)((ga >> 32) & 0x01FFFFFFull) | 0x80000000u; // addr[56:32] | type=2
    i32x8 g1;
    g1[0] = (int)cfg;
    g1[1] = (tdim0 & 0xFFFF) << 16;                               // tensor_dim0 lo
    g1[2] = ((tdim0 >> 16) & 0xFFFF) | ((tdim1 & 0xFFFF) << 16);  // dim0 hi | dim1 lo
    g1[3] = ((tdim1 >> 16) & 0xFFFF) | (tile_d0 << 16);           // dim1 hi | tile_dim0
    g1[4] = tile_d1 & 0xFFFF;                                     // tile_dim1
    g1[5] = (int)(stride & 0xFFFFFFFFll);                         // dim0_stride lo
    g1[6] = (int)((stride >> 32) & 0xFFFFll);                     // dim0_stride hi
    g1[7] = 0;
    i32x4 z4 = {0, 0, 0, 0};
#if defined(__clang_major__) && (__clang_major__ >= 23)
    i32x8 z8 = {0, 0, 0, 0, 0, 0, 0, 0};
    __builtin_amdgcn_tensor_load_to_lds(g0, g1, z4, z4, z8, 0);
#else
    __builtin_amdgcn_tensor_load_to_lds(g0, g1, z4, z4, 0);
#endif
}
#else
#define USE_TDM 0
#endif

// cfg words: data_size=2B | pad_enable | pad_interval | pad_amount
#define TDM_CFG_GEMM 0x06D10000u   // 32-elem rows, pad 8 bf16 -> 40-bf16 stride
#define TDM_CFG_ATTK 0x07110000u   // 64-elem rows, pad 8 bf16 -> 72-bf16 stride

// ---------------------------------------------------------------------------
// GEMM:  C[M,N] = A[M,K] * W[N,K]^T + bias[N]   (optional ReLU, f32/bf16 out)
// A, W bf16 row-major.  128x128x32 tiles, 256 thr = 8 waves (2x4).
// TDM double-buffered pipeline: wave0 issues next tile pair after the
// barrier (previous buffer provably drained), computes overlap with DMA.
// ---------------------------------------------------------------------------
#define BM 128
#define BN 128
#define BK 32
#define LDS_STRIDE_U32 20   // 40 bf16 per row, rows 16B aligned
#define TILE_U32 (BM * LDS_STRIDE_U32)

__device__ __forceinline__ void mma_tiles(const unsigned* tA, const unsigned* tB,
                                          int wm, int wn, int arow, int halfkW,
                                          v8f acc[4][2])
{
    AF a[4], b[2];
#pragma unroll
    for (int mt = 0; mt < 4; ++mt) {
        const unsigned* base = tA + (wm * 64 + mt * 16 + arow) * LDS_STRIDE_U32;
        a[mt].q[0] = *(const u32x4*)(base + halfkW);
        a[mt].q[1] = *(const u32x4*)(base + halfkW + 8);
    }
#pragma unroll
    for (int nt = 0; nt < 2; ++nt) {
        const unsigned* base = tB + (wn * 32 + nt * 16 + arow) * LDS_STRIDE_U32;
        b[nt].q[0] = *(const u32x4*)(base + halfkW);
        b[nt].q[1] = *(const u32x4*)(base + halfkW + 8);
    }
#pragma unroll
    for (int mt = 0; mt < 4; ++mt)
#pragma unroll
        for (int nt = 0; nt < 2; ++nt)
            acc[mt][nt] = __builtin_amdgcn_wmma_f32_16x16x32_bf16(
                false, a[mt].v, false, b[nt].v, (short)0, acc[mt][nt], false, false);
}

__global__ __launch_bounds__(256)
void wmma_gemm(const unsigned short* __restrict__ A,
               const unsigned short* __restrict__ W,
               const float* __restrict__ bias, void* __restrict__ Cv,
               int M, int N, int K, int relu, int out_bf16)
{
    // [buf][A=0/B=1][tile]
    __shared__ __attribute__((aligned(16))) unsigned ltile[4][TILE_U32];

    const int tid  = threadIdx.x;
    const int lane = tid & 31;
    const int wave = tid >> 5;
    const int wm   = wave >> 2;
    const int wn   = wave & 3;
    const int rowBase = blockIdx.y * BM;
    const int colBase = blockIdx.x * BN;

    const int arow   = lane & 15;
    const int halfkW = (lane < 16) ? 0 : 4;
    const int rowAdd = (lane < 16) ? 0 : 8;

    v8f acc[4][2];
#pragma unroll
    for (int mt = 0; mt < 4; ++mt)
#pragma unroll
        for (int nt = 0; nt < 2; ++nt) acc[mt][nt] = v8f_zero();

#if USE_TDM
    const unsigned offA0 = (unsigned)(uintptr_t)(void*)ltile[0];
    const unsigned offB0 = (unsigned)(uintptr_t)(void*)ltile[1];
    const unsigned offA1 = (unsigned)(uintptr_t)(void*)ltile[2];
    const unsigned offB1 = (unsigned)(uintptr_t)(void*)ltile[3];
    const unsigned short* Ab = A + (long)rowBase * K;
    const unsigned short* Wb = W + (long)colBase * K;

    if (wave == 0) {
        tdm_load_tile(offA0, Ab, K, M - rowBase, BK, BM, K, TDM_CFG_GEMM);
        tdm_load_tile(offB0, Wb, K, N - colBase, BK, BN, K, TDM_CFG_GEMM);
    }
    int cur = 0;
    for (int k0 = 0; k0 < K; k0 += BK) {
        if (wave == 0) __builtin_amdgcn_s_wait_tensorcnt(0);
        __syncthreads();                       // cur tile ready; prev buffer drained
        int kn = k0 + BK;
        if (kn < K && wave == 0) {             // prefetch next pair into other buffer
            tdm_load_tile(cur ? offA0 : offA1, Ab + kn, K, M - rowBase, BK, BM, K, TDM_CFG_GEMM);
            tdm_load_tile(cur ? offB0 : offB1, Wb + kn, K, N - colBase, BK, BN, K, TDM_CFG_GEMM);
        }
        mma_tiles(ltile[cur * 2], ltile[cur * 2 + 1], wm, wn, arow, halfkW, acc);
        cur ^= 1;
    }
#else
    for (int k0 = 0; k0 < K; k0 += BK) {
        __syncthreads();
        for (int p = tid; p < BM * 16; p += 256) {
            int r = p >> 4, w = p & 15;
            int gr = rowBase + r, gn = colBase + r;
            unsigned av = 0, bv = 0;
            if (gr < M) av = ((const unsigned*)(A + (long)gr * K + k0))[w];
            if (gn < N) bv = ((const unsigned*)(W + (long)gn * K + k0))[w];
            ltile[0][r * LDS_STRIDE_U32 + w] = av;
            ltile[1][r * LDS_STRIDE_U32 + w] = bv;
        }
        __syncthreads();
        mma_tiles(ltile[0], ltile[1], wm, wn, arow, halfkW, acc);
    }
#endif

    // ---- epilogue: bias (+ReLU), f32 or bf16 stores, guarded
#pragma unroll
    for (int nt = 0; nt < 2; ++nt) {
        int gcol = colBase + wn * 32 + nt * 16 + arow;
        float bv = (gcol < N) ? bias[gcol] : 0.f;
#pragma unroll
        for (int mt = 0; mt < 4; ++mt) {
#pragma unroll
            for (int v = 0; v < 8; ++v) {
                int grow = rowBase + wm * 64 + mt * 16 + v + rowAdd;
                if (grow < M && gcol < N) {
                    float x = acc[mt][nt][v] + bv;
                    if (relu) x = fmaxf(x, 0.f);
                    if (out_bf16)
                        ((unsigned short*)Cv)[(long)grow * N + gcol] = bf16_bits(x);
                    else
                        ((float*)Cv)[(long)grow * N + gcol] = x;
                }
            }
        }
    }
}

// ---------------------------------------------------------------------------
// Fused multi-head attention (flash-style online softmax), bf16 in/out.
// One workgroup (256 thr = 8 waves) per (b,h).  S=512, HD=64.
// K tile staged by TDM (512x64, stride B*3D) while all threads stage V^T.
// ---------------------------------------------------------------------------
#define ATT_3D     3072
#define ATT_D      1024
#define ATT_SB3D   49152           // B*3D
#define ATT_SBD    16384           // B*D
#define K_STRIDE   36              // u32 per K row (72 bf16)
#define V_STRIDE   260             // u32 per V^T row (520 bf16)
#define ATT_LDS_U32 (512*K_STRIDE + 64*V_STRIDE + 8*256)
#define ATT_SCALE  0.125f

__global__ __launch_bounds__(256)
void attn_fused(const unsigned short* __restrict__ qkv, unsigned short* __restrict__ o)
{
    extern __shared__ __attribute__((aligned(16))) unsigned smem[];
    unsigned* Kl = smem;
    unsigned* Vl = Kl + 512 * K_STRIDE;
    unsigned* Pl = Vl + 64 * V_STRIDE;

    const int bh  = blockIdx.x;
    const int b   = bh >> 4;
    const int h   = bh & 15;
    const int tid = threadIdx.x;
    const int lane = tid & 31;
    const int wave = tid >> 5;
    const long base = (long)b * ATT_3D + h * 64;

#if USE_TDM
    if (wave == 0) {
        tdm_load_tile((unsigned)(uintptr_t)(void*)Kl, qkv + base + ATT_D,
                      64, 512, 64, 512, (long)ATT_SB3D, TDM_CFG_ATTK);
    }
#else
    {
        const unsigned* q32 = (const unsigned*)qkv;
        for (int i = tid; i < 512 * 32; i += 256) {
            int s = i >> 5, w = i & 31;
            Kl[s * K_STRIDE + w] = q32[(((long)s * ATT_SB3D + base + ATT_D) >> 1) + w];
        }
    }
#endif
    // ---- stage V transposed: Vt[d][s']  (overlaps with the K TDM)
    for (int i = tid; i < 64 * 256; i += 256) {
        int d = i >> 8, sp = (i & 255) << 1;
        unsigned short v0 = qkv[(long)sp       * ATT_SB3D + base + 2 * ATT_D + d];
        unsigned short v1 = qkv[(long)(sp + 1) * ATT_SB3D + base + 2 * ATT_D + d];
        Vl[d * V_STRIDE + (sp >> 1)] = (unsigned)v0 | ((unsigned)v1 << 16);
    }
#if USE_TDM
    if (wave == 0) __builtin_amdgcn_s_wait_tensorcnt(0);
#endif
    __syncthreads();

    const int arow   = lane & 15;
    const int halfE  = (lane < 16) ? 0 : 8;
    const int halfkW = (lane < 16) ? 0 : 4;
    const int rowAdd = (lane < 16) ? 0 : 8;
    unsigned* Pw = Pl + wave * 256;

    for (int qt = wave; qt < 32; qt += 8) {
        AF qa[2];
        {
            const unsigned short* qp = qkv + (long)(qt * 16 + arow) * ATT_SB3D + base;
#pragma unroll
            for (int ks = 0; ks < 2; ++ks) {
                qa[ks].q[0] = *(const u32x4*)(qp + ks * 32 + halfE);
                qa[ks].q[1] = *(const u32x4*)(qp + ks * 32 + halfE + 16);
            }
        }

        float m[8], lsum[8];
        v8f oc[4];
#pragma unroll
        for (int v = 0; v < 8; ++v) { m[v] = -1e30f; lsum[v] = 0.f; }
#pragma unroll
        for (int dt = 0; dt < 4; ++dt) oc[dt] = v8f_zero();

        for (int kt = 0; kt < 16; ++kt) {
            v8f sc[2];
            sc[0] = v8f_zero(); sc[1] = v8f_zero();
#pragma unroll
            for (int nt = 0; nt < 2; ++nt) {
                const unsigned* kb = Kl + (kt * 32 + nt * 16 + arow) * K_STRIDE;
#pragma unroll
                for (int ks = 0; ks < 2; ++ks) {
                    AF kf;
                    kf.q[0] = *(const u32x4*)(kb + ks * 16 + halfkW);
                    kf.q[1] = *(const u32x4*)(kb + ks * 16 + halfkW + 8);
                    sc[nt] = __builtin_amdgcn_wmma_f32_16x16x32_bf16(
                        false, qa[ks].v, false, kf.v, (short)0, sc[nt], false, false);
                }
            }
#pragma unroll
            for (int v = 0; v < 8; ++v) {
                float s0 = sc[0][v] * ATT_SCALE, s1 = sc[1][v] * ATT_SCALE;
                float r = fmaxf(s0, s1);
                r = fmaxf(r, __shfl_xor(r, 1, 32));
                r = fmaxf(r, __shfl_xor(r, 2, 32));
                r = fmaxf(r, __shfl_xor(r, 4, 32));
                r = fmaxf(r, __shfl_xor(r, 8, 32));
                float nm    = fmaxf(m[v], r);
                float alpha = __expf(m[v] - nm);
                float p0 = __expf(s0 - nm);
                float p1 = __expf(s1 - nm);
                float rs = p0 + p1;
                rs += __shfl_xor(rs, 1, 32);
                rs += __shfl_xor(rs, 2, 32);
                rs += __shfl_xor(rs, 4, 32);
                rs += __shfl_xor(rs, 8, 32);
                lsum[v] = lsum[v] * alpha + rs;
                m[v]    = nm;
#pragma unroll
                for (int dt = 0; dt < 4; ++dt) oc[dt][v] *= alpha;
                unsigned short* Pr = (unsigned short*)Pw + (v + rowAdd) * 32;
                Pr[arow]      = bf16_bits(p0);
                Pr[16 + arow] = bf16_bits(p1);
            }
            AF pa;
            {
                const unsigned* pb = Pw + arow * 16;
                pa.q[0] = *(const u32x4*)(pb + halfkW);
                pa.q[1] = *(const u32x4*)(pb + halfkW + 8);
            }
#pragma unroll
            for (int dt = 0; dt < 4; ++dt) {
                const unsigned* vb = Vl + (dt * 16 + arow) * V_STRIDE + kt * 16;
                AF vf;
                vf.q[0] = *(const u32x4*)(vb + halfkW);
                vf.q[1] = *(const u32x4*)(vb + halfkW + 8);
                oc[dt] = __builtin_amdgcn_wmma_f32_16x16x32_bf16(
                    false, pa.v, false, vf.v, (short)0, oc[dt], false, false);
            }
        }
#pragma unroll
        for (int v = 0; v < 8; ++v) {
            float inv = 1.0f / lsum[v];
            long srow = qt * 16 + v + rowAdd;
            unsigned short* op = o + srow * ATT_SBD + (long)b * ATT_D + h * 64 + arow;
#pragma unroll
            for (int dt = 0; dt < 4; ++dt)
                op[dt * 16] = bf16_bits(oc[dt][v] * inv);
        }
    }
}

// ---------------------------------------------------------------------------
// y = LayerNorm(x + t)*s + b ; writes f32 and bf16 copies.  D=1024, 1 row/block
// ---------------------------------------------------------------------------
__global__ __launch_bounds__(256)
void add_ln(const float* __restrict__ x, const float* __restrict__ t,
            const float* __restrict__ s, const float* __restrict__ bn,
            float* __restrict__ y, unsigned short* __restrict__ ybf)
{
    __shared__ float red[256];
    const int row = blockIdx.x;
    const int tid = threadIdx.x;
    const float* xr = x + (long)row * 1024;
    const float* tr = t + (long)row * 1024;

    float v[4];
    float acc = 0.f;
#pragma unroll
    for (int i = 0; i < 4; ++i) { v[i] = xr[tid + i * 256] + tr[tid + i * 256]; acc += v[i]; }
    red[tid] = acc; __syncthreads();
    for (int o = 128; o > 0; o >>= 1) { if (tid < o) red[tid] += red[tid + o]; __syncthreads(); }
    const float mean = red[0] * (1.0f / 1024.0f);
    __syncthreads();

    acc = 0.f;
#pragma unroll
    for (int i = 0; i < 4; ++i) { float d = v[i] - mean; acc += d * d; }
    red[tid] = acc; __syncthreads();
    for (int o = 128; o > 0; o >>= 1) { if (tid < o) red[tid] += red[tid + o]; __syncthreads(); }
    const float rstd = rsqrtf(red[0] * (1.0f / 1024.0f) + 1e-5f);

    float*          yr = y   + (long)row * 1024;
    unsigned short* yb = ybf + (long)row * 1024;
#pragma unroll
    for (int i = 0; i < 4; ++i) {
        int c = tid + i * 256;
        float out = (v[i] - mean) * rstd * s[c] + bn[c];
        yr[c] = out;
        yb[c] = bf16_bits(out);
    }
}

// fp32 -> bf16 bulk convert (4 elems/thread)
__global__ void f32_to_bf16_kernel(const float* __restrict__ in,
                                   unsigned short* __restrict__ out, int n4)
{
    int g = blockIdx.x * 256 + threadIdx.x;
    if (g < n4) {
        f32x4 v = *(const f32x4*)(in + (long)g * 4);
        __builtin_prefetch(in + (long)g * 4 + 4096);
        unsigned* o = (unsigned*)out;
        o[(long)g * 2]     = pack2(v[0], v[1]);
        o[(long)g * 2 + 1] = pack2(v[2], v[3]);
    }
}

// rep[b,d] = mean_s h[s,b,d]
__global__ void mean_s_kernel(const float* __restrict__ hbuf, float* __restrict__ rep)
{
    int g = blockIdx.x * 256 + threadIdx.x;
    float acc = 0.f;
    for (int s = 0; s < 512; ++s) acc += hbuf[(long)s * 16384 + g];
    rep[g] = acc * (1.0f / 512.0f);
}

// gating: logits = rep @ Wg^T + bg ; softmax ; argmax (first-max)
__global__ void gating_kernel(const float* __restrict__ rep, const float* __restrict__ Wg,
                              const float* __restrict__ bg, float* __restrict__ gw_out,
                              int* __restrict__ idx_out, int* __restrict__ eidx)
{
    __shared__ float lg[16][8];
    const int t = threadIdx.x;
    if (t < 128) {
        int b = t >> 3, e = t & 7;
        float acc = bg[e];
        const float* rp = rep + b * 1024;
        const float* wp = Wg + e * 1024;
        for (int d = 0; d < 1024; ++d) acc += rp[d] * wp[d];
        lg[b][e] = acc;
    }
    __syncthreads();
    if (t < 16) {
        float mx = lg[t][0];
#pragma unroll
        for (int e = 1; e < 8; ++e) mx = fmaxf(mx, lg[t][e]);
        float w[8], sum = 0.f;
#pragma unroll
        for (int e = 0; e < 8; ++e) { w[e] = __expf(lg[t][e] - mx); sum += w[e]; }
        float inv = 1.0f / sum;
        int best = 0; float bw = -1.f;
#pragma unroll
        for (int e = 0; e < 8; ++e) {
            float ww = w[e] * inv;
            gw_out[t * 8 + e] = ww;
            if (ww > bw) { bw = ww; best = e; }
        }
        eidx[t]    = best;
        idx_out[t] = best;
    }
}

// expert_out[b,o] = rep[b,:] . We[e_b, o, :] + be[e_b, o]  (bf16 out for head)
__global__ void expert_kernel(const float* __restrict__ rep, const float* __restrict__ We,
                              const float* __restrict__ be, const int* __restrict__ eidx,
                              unsigned short* __restrict__ eout)
{
    int g = blockIdx.x * 256 + threadIdx.x;
    int b = g >> 10, o = g & 1023;
    int e = eidx[b];
    const float* rp = rep + b * 1024;
    const float* wp = We + (long)e * 1024 * 1024 + (long)o * 1024;
    float acc = be[e * 1024 + o];
    for (int d = 0; d < 1024; ++d) acc += rp[d] * wp[d];
    eout[g] = bf16_bits(acc);
}

// ---------------------------------------------------------------------------
// Orchestration
// ---------------------------------------------------------------------------
static inline int cdiv(int a, int b) { return (a + b - 1) / b; }

extern "C" void kernel_launch(void* const* d_in, const int* in_sizes, int n_in,
                              void* d_out, int out_size, void* d_ws, size_t ws_size,
                              hipStream_t stream)
{
    (void)in_sizes; (void)n_in; (void)out_size; (void)ws_size;

    const float* x     = (const float*)d_in[0];
    const float* Wqkv  = (const float*)d_in[1];
    const float* bqkv  = (const float*)d_in[2];
    const float* Wo    = (const float*)d_in[3];
    const float* bo    = (const float*)d_in[4];
    const float* ln1s  = (const float*)d_in[5];
    const float* ln1b  = (const float*)d_in[6];
    const float* W1    = (const float*)d_in[7];
    const float* b1    = (const float*)d_in[8];
    const float* W2    = (const float*)d_in[9];
    const float* b2    = (const float*)d_in[10];
    const float* ln2s  = (const float*)d_in[11];
    const float* ln2b  = (const float*)d_in[12];
    const float* Wg    = (const float*)d_in[13];
    const float* bg    = (const float*)d_in[14];
    const float* We    = (const float*)d_in[15];
    const float* be    = (const float*)d_in[16];
    const float* Wh    = (const float*)d_in[17];
    const float* bh    = (const float*)d_in[18];

    char* base = (char*)d_ws;
    float*          h      = (float*)(base + 0);                  //  32 MB f32
    float*          proj   = (float*)(base + 33554432);           //  32 MB f32
    unsigned short* h_bf   = (unsigned short*)(base + 67108864);  //  16 MB bf16
    unsigned short* qkv_bf = (unsigned short*)(base + 83886080);  //  48 MB bf16 (ff1 reuse)
    unsigned short* ao_bf  = (unsigned short*)(base + 134217728); //  16 MB bf16
    unsigned short* wbuf   = (unsigned short*)(base + 150994944); //   6 MB bf16 (layer weights)
    float*          rep    = (float*)(base + 157286400);
    unsigned short* eout   = (unsigned short*)(base + 157351936);
    int*            eidx   = (int*)(base + 157384704);
    // Wh bf16 overlays proj/h_bf/qkv_bf (all dead by then): 98 MB needed
    unsigned short* wh_bf  = (unsigned short*)(base + 33554432);

    hipMemcpyAsync(h, x, (size_t)8388608 * sizeof(float), hipMemcpyDeviceToDevice, stream);
    f32_to_bf16_kernel<<<dim3(cdiv(2097152, 256)), 256, 0, stream>>>(x, h_bf, 2097152);

    for (int i = 0; i < 4; ++i) {
        const float* Wq  = Wqkv + (size_t)i * 3072 * 1024;
        const float* bq  = bqkv + (size_t)i * 3072;
        const float* Woi = Wo   + (size_t)i * 1024 * 1024;
        const float* boi = bo   + (size_t)i * 1024;
        const float* s1  = ln1s + (size_t)i * 1024;
        const float* n1  = ln1b + (size_t)i * 1024;
        const float* W1i = W1   + (size_t)i * 2048 * 1024;
        const float* b1i = b1   + (size_t)i * 2048;
        const float* W2i = W2   + (size_t)i * 1024 * 2048;
        const float* b2i = b2   + (size_t)i * 1024;
        const float* s2  = ln2s + (size_t)i * 1024;
        const float* n2  = ln2b + (size_t)i * 1024;

        // qkv = h @ Wqkv^T + bqkv  (bf16 out)
        f32_to_bf16_kernel<<<dim3(cdiv(786432, 256)), 256, 0, stream>>>(Wq, wbuf, 786432);
        wmma_gemm<<<dim3(24, 64), 256, 0, stream>>>(h_bf, wbuf, bq, qkv_bf,
                                                    8192, 3072, 1024, 0, 1);
        // fused attention -> ao (bf16)
        attn_fused<<<dim3(256), 256, ATT_LDS_U32 * sizeof(unsigned), stream>>>(qkv_bf, ao_bf);
        // proj = ao @ Wo^T + bo  (f32 out)
        f32_to_bf16_kernel<<<dim3(cdiv(262144, 256)), 256, 0, stream>>>(Woi, wbuf, 262144);
        wmma_gemm<<<dim3(8, 64), 256, 0, stream>>>(ao_bf, wbuf, boi, proj,
                                                   8192, 1024, 1024, 0, 0);
        // h = LN(h + proj)
        add_ln<<<dim3(8192), 256, 0, stream>>>(h, proj, s1, n1, h, h_bf);
        // ff1 = relu(h @ W1^T + b1)  (bf16 out, reuses qkv buffer)
        f32_to_bf16_kernel<<<dim3(cdiv(524288, 256)), 256, 0, stream>>>(W1i, wbuf, 524288);
        wmma_gemm<<<dim3(16, 64), 256, 0, stream>>>(h_bf, wbuf, b1i, qkv_bf,
                                                    8192, 2048, 1024, 1, 1);
        // proj = ff1 @ W2^T + b2  (f32 out)
        f32_to_bf16_kernel<<<dim3(cdiv(524288, 256)), 256, 0, stream>>>(W2i, wbuf, 524288);
        wmma_gemm<<<dim3(8, 64), 256, 0, stream>>>(qkv_bf, wbuf, b2i, proj,
                                                   8192, 1024, 2048, 0, 0);
        // h = LN(h + proj)
        add_ln<<<dim3(8192), 256, 0, stream>>>(h, proj, s2, n2, h, h_bf);
    }

    mean_s_kernel<<<dim3(64), 256, 0, stream>>>(h, rep);

    float* out_logits = (float*)d_out;                  // 16*50257
    float* out_gw     = (float*)d_out + 804112;         // 16*8
    int*   out_idx    = (int*)((float*)d_out + 804240); // 16

    gating_kernel<<<dim3(1), 128, 0, stream>>>(rep, Wg, bg, out_gw, out_idx, eidx);
    expert_kernel<<<dim3(64), 256, 0, stream>>>(rep, We, be, eidx, eout);

    // logits = expert_out @ Wh^T + bh  (f32 out)
    f32_to_bf16_kernel<<<dim3(cdiv(12865792, 256)), 256, 0, stream>>>(Wh, wh_bf, 12865792);
    wmma_gemm<<<dim3(393, 1), 256, 0, stream>>>(eout, wh_bf, bh, out_logits,
                                                16, 50257, 1024, 0, 0);
}
